// TGAT_52828097741151
// MI455X (gfx1250) — matrix operand order repaired and checked by reading, not compiled, not run on hardware
//
#include <hip/hip_runtime.h>
#include <math.h>

#define NNODES 50000
#define KNBR   16
#define INFEAT 128
#define HID    128
#define T2VD   64
#define FDIM   192   // INFEAT + T2VD

typedef __attribute__((ext_vector_type(16))) __bf16 v16bf;
typedef __attribute__((ext_vector_type(8)))  float  v8f;
typedef __attribute__((ext_vector_type(2)))  float  v2f;
typedef __attribute__((ext_vector_type(2)))  __bf16 v2bf;

// Fragment-major sizes (bf16 elements)
#define AFRAG_ELEMS (6 * 32 * 16)      // 16x192 A-matrix, frag-major  (3072)
#define AGG_ELEMS   (4 * 32 * 16)      // 16x128 A-matrix, frag-major  (2048)
#define WKV_ELEMS   (6 * 8 * 32 * 16)  // 192x128 B-matrix, frag-major (24576)

// Packed f32x2 -> bf16x2 (clang lowers to a single v_cvt_pk_bf16_f32)
__device__ __forceinline__ unsigned pack2(float lo, float hi) {
    v2f a = {lo, hi};
    v2bf b = __builtin_convertvector(a, v2bf);
    union { v2bf v; unsigned u; } c; c.v = b;
    return c.u;
}

// A-matrix (16x32 bf16, ISA 7.12.2): element (m, f) -> frag-major index.
__device__ __forceinline__ int a_dst(int m, int f) {
    int t = f >> 5, r = f & 31;
    int lane = m + (((r >> 3) & 1) << 4);
    int off  = ((r >> 4) << 3) | (r & 7);
    return (t * 32 + lane) * 16 + off;
}

// Stage a row-major fp32 weight [ksteps*32 x 128] into B-fragment-major bf16 LDS.
// 4 consecutive k-rows of one column are contiguous in the frag layout.
__device__ __forceinline__ void stage_b4(const float* __restrict__ W, __bf16* dst, int ksteps, int tid) {
    const int total = ksteps * 32 * 128 / 4;
    for (int e = tid; e < total; e += 256) {
        int col  = e & 127;
        int krow = (e >> 7) << 2;
        const float* p = W + (size_t)krow * HID + col;
        uint2 uv;
        uv.x = pack2(p[0],       p[HID]);
        uv.y = pack2(p[2 * HID], p[3 * HID]);
        int t    = krow >> 5;
        int hi   = (krow >> 3) & 1;
        int voff = ((krow >> 4) & 1) * 8 + (krow & 7);   // multiple of 4
        int d = (((t * 8 + (col >> 4)) * 32) + (col & 15) + hi * 16) * 16 + voff;
        *(uint2*)&dst[d] = uv;
    }
}

__device__ __forceinline__ v8f wmma_bf16(v16bf a, v16bf b, v8f c) {
    return __builtin_amdgcn_wmma_f32_16x16x32_bf16(false, a, false, b, (short)0, c, false, false);
}

__global__ __launch_bounds__(256) void tgat_attn_kernel(
    const float* __restrict__ x,  const float* __restrict__ ts, const int* __restrict__ idx,
    const float* __restrict__ w0p, const float* __restrict__ b0p,
    const float* __restrict__ Wt, const float* __restrict__ Bt,
    const float* __restrict__ Wq, const float* __restrict__ Wk,
    const float* __restrict__ Wv, const float* __restrict__ Wo,
    const float* __restrict__ bo, float* __restrict__ out)
{
    // 158 KB total -> two workgroups per 320 KB WGP
    __shared__ __attribute__((aligned(32))) __bf16 sWa[WKV_ELEMS];      // 48 KB: Wq, then Wk, then Wo
    __shared__ __attribute__((aligned(32))) __bf16 sWv[WKV_ELEMS];      // 48 KB: Wv
    __shared__ __attribute__((aligned(32))) __bf16 sHn[8][AFRAG_ELEMS]; // 48 KB, per-wave h_nbr
    __shared__ __attribute__((aligned(32))) __bf16 sHs[AFRAG_ELEMS];    //  6 KB
    __shared__ __attribute__((aligned(32))) __bf16 sAgg[AGG_ELEMS];     //  4 KB
    __shared__ __attribute__((aligned(16))) __bf16 sQ[16][128];         //  4 KB, pre-scaled q

    const int tid  = threadIdx.x;
    const int lane = tid & 31;
    const int wave = tid >> 5;
    const int base = blockIdx.x * 16;

    const float w0v = w0p[0];
    const float b0v = b0p[0];

    // ---- Phase 0: stage Wq (B-frag-major) + h_self (A-frag-major) ----
    stage_b4(Wq, sWa, 6, tid);
    for (int e = tid; e < 16 * FDIM; e += 256) {
        int m = e / FDIM, f = e - m * FDIM;
        int row = base + m; if (row >= NNODES) row = NNODES - 1;
        float val;
        if (f < INFEAT)          val = x[(size_t)row * INFEAT + f];
        else if (f == INFEAT)    val = b0v;                  // lin term at t=0
        else                     val = sinf(Bt[f - INFEAT - 1]);
        sHs[a_dst(m, f)] = (__bf16)val;
    }
    __syncthreads();

    // ---- Phase 1: Q = Hself @ Wq (wave w -> cols 16w..16w+15), stored scaled ----
    {
        v8f acc = {0.f, 0.f, 0.f, 0.f, 0.f, 0.f, 0.f, 0.f};
#pragma unroll
        for (int t = 0; t < 6; ++t) {
            v16bf a = *(const v16bf*)&sHs[(t * 32 + lane) * 16];
            v16bf b = *(const v16bf*)&sWa[((t * 8 + wave) * 32 + lane) * 16];
            acc = wmma_bf16(a, b, acc);
        }
        const int n = lane & 15, hi = lane >> 4;
#pragma unroll
        for (int r = 0; r < 8; ++r)
            sQ[r + 8 * hi][16 * wave + n] = (__bf16)(acc[r] * 0.08838834764831845f);
    }
    __syncthreads();

    // ---- Phase 2: each wave handles 2 nodes ----
    for (int it = 0; it < 2; ++it) {
        const int nrow = wave * 2 + it;
        int node = base + nrow; if (node >= NNODES) node = NNODES - 1;

        if (it == 0) {                // overwrite Wq with Wk; stage Wv
            stage_b4(Wk, sWa, 6, tid);
            stage_b4(Wv, sWv, 6, tid);
        }

        int   nbr = 0; float tvl = 0.f;
        if (lane < KNBR) {
            nbr = idx[node * KNBR + lane];
            tvl = ts[node * KNBR + lane];
        }
        // stage gathered x rows (float4 -> 4 contiguous bf16 in frag layout)
        for (int e = lane; e < KNBR * (INFEAT / 4); e += 32) {
            int m = e >> 5, q = e & 31;                  // f = q*4
            int row = __shfl(nbr, m, 32);
            const float4 vv = *(const float4*)&x[(size_t)row * INFEAT + q * 4];
            uint2 uv; uv.x = pack2(vv.x, vv.y); uv.y = pack2(vv.z, vv.w);
            int t      = q >> 3;
            int lane_d = m + (((q >> 1) & 1) << 4);
            int off    = ((q >> 2) & 1) * 8 + (q & 1) * 4;
            *(uint2*)&sHn[wave][(t * 32 + lane_d) * 16 + off] = uv;
        }
        // stage time2vec(ts)
        for (int e = lane; e < KNBR * T2VD; e += 32) {
            int m = e >> 6, j = e & 63;
            float t = __shfl(tvl, m, 32);
            float val = (j == 0) ? (t * w0v + b0v) : sinf(t * Wt[j - 1] + Bt[j - 1]);
            sHn[wave][a_dst(m, INFEAT + j)] = (__bf16)val;
        }
        __syncthreads();  // uniform across block

        const int n = lane & 15, hi = lane >> 4;

        // K-projection
        v8f accK[8];
#pragma unroll
        for (int j = 0; j < 8; ++j) accK[j] = (v8f){0.f, 0.f, 0.f, 0.f, 0.f, 0.f, 0.f, 0.f};
#pragma unroll
        for (int t = 0; t < 6; ++t) {
            v16bf a = *(const v16bf*)&sHn[wave][(t * 32 + lane) * 16];
#pragma unroll
            for (int j = 0; j < 8; ++j) {
                v16bf b = *(const v16bf*)&sWa[((t * 8 + j) * 32 + lane) * 16];
                accK[j] = wmma_bf16(a, b, accK[j]);
            }
        }

        // scores[m] = q . k[m]  (q pre-scaled). D layout: lane holds (m = r+8*hi, n)
        float qv[8];
#pragma unroll
        for (int j = 0; j < 8; ++j) qv[j] = (float)sQ[nrow][16 * j + n];
        float sc[8];
#pragma unroll
        for (int r = 0; r < 8; ++r) {
            float p = 0.f;
#pragma unroll
            for (int j = 0; j < 8; ++j) p += accK[j][r] * qv[j];
            p += __shfl_xor(p, 1, 32);
            p += __shfl_xor(p, 2, 32);
            p += __shfl_xor(p, 4, 32);
            p += __shfl_xor(p, 8, 32);
            sc[r] = p;
        }

        // softmax over 16 neighbors
        float mx = sc[0];
#pragma unroll
        for (int r = 1; r < 8; ++r) mx = fmaxf(mx, sc[r]);
        mx = fmaxf(mx, __shfl_xor(mx, 16, 32));
        float ex[8], se = 0.f;
#pragma unroll
        for (int r = 0; r < 8; ++r) { ex[r] = expf(sc[r] - mx); se += ex[r]; }
        se += __shfl_xor(se, 16, 32);
        float inv = 1.f / se;
#pragma unroll
        for (int r = 0; r < 8; ++r) ex[r] *= inv;

        // V-projection
        v8f accV[8];
#pragma unroll
        for (int j = 0; j < 8; ++j) accV[j] = (v8f){0.f, 0.f, 0.f, 0.f, 0.f, 0.f, 0.f, 0.f};
#pragma unroll
        for (int t = 0; t < 6; ++t) {
            v16bf a = *(const v16bf*)&sHn[wave][(t * 32 + lane) * 16];
#pragma unroll
            for (int j = 0; j < 8; ++j) {
                v16bf b = *(const v16bf*)&sWv[((t * 8 + j) * 32 + lane) * 16];
                accV[j] = wmma_bf16(a, b, accV[j]);
            }
        }

        // agg[h] = sum_m attn[m] * v[m][h]; write into sAgg (A-frag-major)
#pragma unroll
        for (int j = 0; j < 8; ++j) {
            float p = 0.f;
#pragma unroll
            for (int r = 0; r < 8; ++r) p += ex[r] * accV[j][r];
            p += __shfl_xor(p, 16, 32);
            if (hi == 0) sAgg[a_dst(nrow, 16 * j + n)] = (__bf16)p;
        }
        __syncthreads();
    }

    // ---- Phase 3: stage Wo over sWa, then out = relu(Agg @ Wo + bo) ----
    stage_b4(Wo, sWa, 4, tid);
    __syncthreads();
    {
        v8f acc = {0.f, 0.f, 0.f, 0.f, 0.f, 0.f, 0.f, 0.f};
#pragma unroll
        for (int t = 0; t < 4; ++t) {
            v16bf a = *(const v16bf*)&sAgg[(t * 32 + lane) * 16];
            v16bf b = *(const v16bf*)&sWa[((t * 8 + wave) * 32 + lane) * 16];
            acc = wmma_bf16(a, b, acc);
        }
        const int n = lane & 15, hi = lane >> 4;
        const float bv = bo[16 * wave + n];
#pragma unroll
        for (int r = 0; r < 8; ++r) {
            int row = base + r + 8 * hi;
            if (row < NNODES) {
                float v = acc[r] + bv;
                out[(size_t)row * HID + 16 * wave + n] = v > 0.f ? v : 0.f;
            }
        }
    }
}

extern "C" void kernel_launch(void* const* d_in, const int* in_sizes, int n_in,
                              void* d_out, int out_size, void* d_ws, size_t ws_size,
                              hipStream_t stream) {
    (void)in_sizes; (void)n_in; (void)out_size; (void)d_ws; (void)ws_size;
    const float* x   = (const float*)d_in[0];
    const float* ts  = (const float*)d_in[1];
    const int*   idx = (const int*)  d_in[2];
    const float* w0  = (const float*)d_in[3];
    const float* b0  = (const float*)d_in[4];
    const float* Wt  = (const float*)d_in[5];
    const float* Bt  = (const float*)d_in[6];
    const float* Wq  = (const float*)d_in[7];
    const float* Wk  = (const float*)d_in[8];
    const float* Wv  = (const float*)d_in[9];
    const float* Wo  = (const float*)d_in[10];
    const float* bo  = (const float*)d_in[11];
    float* out = (float*)d_out;

    dim3 grid((NNODES + 15) / 16);
    dim3 block(256);
    hipLaunchKernelGGL(tgat_attn_kernel, grid, block, 0, stream,
                       x, ts, idx, w0, b0, Wt, Bt, Wq, Wk, Wv, Wo, bo, out);
}
